// LocalTemporalMHRA_6330781794546
// MI455X (gfx1250) — matrix operand order repaired and checked by reading, not compiled, not run on hardware
//
#include <hip/hip_runtime.h>

// ---------------- problem constants ----------------
#define BDIM   8
#define TDIM   16
#define NTOK   196
#define CDIM   768
#define HHEADS 12
#define HD     64
#define NPAD   224          // pad 196 -> 224 (7 K-steps of 32, 14 tiles of 16)
#define OC3    (3*CDIM)     // 2304
#define MROWS  (BDIM*TDIM*NTOK)  // 25088 = 392*64 exactly

typedef __attribute__((ext_vector_type(16))) __bf16    v16bf;
typedef __attribute__((ext_vector_type(8)))  float     v8f;
typedef __attribute__((ext_vector_type(4)))  unsigned  u32x4;
typedef __attribute__((ext_vector_type(4)))  float     f32x4;

union Frag { v16bf v; unsigned u[8]; u32x4 q[2]; };
union Acc  { v8f   v; float    f[8]; };

// fp32 -> bf16 round-to-nearest-even (bit trick), bf16 -> fp32
static __device__ __forceinline__ unsigned short f2bf(float f) {
  unsigned u = __float_as_uint(f);
  u += 0x7FFFu + ((u >> 16) & 1u);
  return (unsigned short)(u >> 16);
}
static __device__ __forceinline__ unsigned pk2(float lo, float hi) {
  return (unsigned)f2bf(lo) | ((unsigned)f2bf(hi) << 16);
}
static __device__ __forceinline__ float bf2f(unsigned short b) {
  return __uint_as_float(((unsigned)b) << 16);
}
static __device__ __forceinline__ u32x4 ldg4(const unsigned short* p) {
  return *(const u32x4*)(const void*)p;
}

// =====================================================================
// Kernel 0: one-shot fp32 -> bf16 conversion (4 elems / thread)
// =====================================================================
__global__ void cvt_kernel(const float* __restrict__ src,
                           unsigned short* __restrict__ dst, int n4) {
  int i = blockIdx.x * blockDim.x + threadIdx.x;
  if (i < n4) {
    f32x4 f = ((const f32x4*)src)[i];
    unsigned lo = pk2(f.x, f.y);
    unsigned hi = pk2(f.z, f.w);
    ((unsigned*)dst)[2 * i]     = lo;
    ((unsigned*)dst)[2 * i + 1] = hi;
  }
}

// =====================================================================
// Kernel 1: QKV projection (bf16 in, bf16 out).
// Wave computes 2 M-tiles x 4 N-tiles (8 accumulators, 8 WMMA per K-step).
// Block = 8 waves (2 M x 4 N) -> 64 rows x 256 cols.
// Epilogue scatters q/k/v into padded [B,H,T,NPAD,HD] layouts.
// =====================================================================
__global__ __launch_bounds__(256)
void qkv_kernel(const unsigned short* __restrict__ xb,
                const unsigned short* __restrict__ wb,
                unsigned short* __restrict__ qb, unsigned short* __restrict__ kb,
                unsigned short* __restrict__ vb) {
  const int lane = threadIdx.x & 31;
  const int wave = threadIdx.x >> 5;
  const int half = lane >> 4;
  const int l16  = lane & 15;
  const int mw   = wave & 1;   // 2 waves in M
  const int nw   = wave >> 1;  // 4 waves in N
  const int rm = blockIdx.y * 64 + mw * 32;   // wave row base (2 tiles)
  const int rn = blockIdx.x * 256 + nw * 64;  // wave col base (4 tiles)

  const unsigned short* ar[2];
  const unsigned short* br[4];
#pragma unroll
  for (int mi = 0; mi < 2; ++mi) ar[mi] = xb + (size_t)(rm + mi * 16 + l16) * CDIM;
#pragma unroll
  for (int ni = 0; ni < 4; ++ni) br[ni] = wb + (size_t)(rn + ni * 16 + l16) * CDIM;

  Acc acc[2][4];
#pragma unroll
  for (int mi = 0; mi < 2; ++mi)
#pragma unroll
    for (int ni = 0; ni < 4; ++ni)
#pragma unroll
      for (int i = 0; i < 8; ++i) acc[mi][ni].f[i] = 0.f;

  for (int k0 = 0; k0 < CDIM; k0 += 32) {
    Frag af[2], bf[4];
#pragma unroll
    for (int mi = 0; mi < 2; ++mi) {      // A: half h holds K {8h..}+{16+8h..}
      af[mi].q[0] = ldg4(ar[mi] + k0 + 8 * half);
      af[mi].q[1] = ldg4(ar[mi] + k0 + 8 * half + 16);
    }
#pragma unroll
    for (int ni = 0; ni < 4; ++ni) {      // B: half h holds K {8h..}+{16+8h..}
      bf[ni].q[0] = ldg4(br[ni] + k0 + 8 * half);
      bf[ni].q[1] = ldg4(br[ni] + k0 + 8 * half + 16);
    }
#pragma unroll
    for (int mi = 0; mi < 2; ++mi)
#pragma unroll
      for (int ni = 0; ni < 4; ++ni)
        acc[mi][ni].v = __builtin_amdgcn_wmma_f32_16x16x32_bf16(
            false, af[mi].v, false, bf[ni].v, (short)0, acc[mi][ni].v, false, false);
  }

  // scatter: D element (m = r + 8*half, n = l16)
#pragma unroll
  for (int mi = 0; mi < 2; ++mi)
#pragma unroll
    for (int r = 0; r < 8; ++r) {
      int m = rm + mi * 16 + r + 8 * half;
      int ntok = m % NTOK;  int bt = m / NTOK;
      int t = bt % TDIM;    int b  = bt / TDIM;
#pragma unroll
      for (int ni = 0; ni < 4; ++ni) {
        int o = rn + ni * 16 + l16;
        int s = o / CDIM;   int rem = o % CDIM;
        int h = rem / HD;   int d = rem % HD;
        size_t bht = (size_t)(b * HHEADS + h) * TDIM + t;
        size_t idx = (bht * NPAD + ntok) * HD + d;
        unsigned short val = f2bf(acc[mi][ni].f[r]);
        if (s == 0)      qb[idx] = val;
        else if (s == 1) kb[idx] = val;
        else             vb[idx] = val;
      }
    }
}

// =====================================================================
// Kernel 2: depthwise 3-tap conv along head-dim + transpose.
// vt[bht][d][n] = w0*v[d-1] + w1*v[d] + w2*v[d+1] + b  (zero for n>=196)
// =====================================================================
__global__ void conv_kernel(const unsigned short* __restrict__ vb,
                            const float* __restrict__ w_dw,
                            const float* __restrict__ b_dw,
                            unsigned short* __restrict__ vt) {
  size_t gid = (size_t)blockIdx.x * blockDim.x + threadIdx.x;
  const size_t total = (size_t)BDIM * HHEADS * TDIM * HD * NPAD;
  if (gid >= total) return;
  int n = (int)(gid % NPAD);  size_t rest = gid / NPAD;
  int d = (int)(rest % HD);   size_t bht  = rest / HD;
  int h = (int)((bht / TDIM) % HHEADS);
  float out = 0.f;
  if (n < NTOK) {
    const unsigned short* vrow = vb + (bht * NPAD + n) * HD;
    float vm = (d > 0)      ? bf2f(vrow[d - 1]) : 0.f;
    float vc =               bf2f(vrow[d]);
    float vp = (d < HD - 1) ? bf2f(vrow[d + 1]) : 0.f;
    out = w_dw[h * 3 + 0] * vm + w_dw[h * 3 + 1] * vc + w_dw[h * 3 + 2] * vp + b_dw[h];
  }
  vt[(bht * HD + d) * NPAD + n] = f2bf(out);
}

// =====================================================================
// Kernel 3: attention per (b,h,t).  8 waves, wave owns 16-query row-blocks.
// =====================================================================
__global__ __launch_bounds__(256)
void attn_kernel(const unsigned short* __restrict__ qb,
                 const unsigned short* __restrict__ kb,
                 const unsigned short* __restrict__ vt,
                 unsigned short* __restrict__ obuf) {
  __shared__ unsigned short lds_p[8][16 * NPAD];  // 57,344 B

  const int lane = threadIdx.x & 31;
  const int wave = threadIdx.x >> 5;
  const int half = lane >> 4;
  const int l16  = lane & 15;
  const size_t bht = blockIdx.x;
  const int t  = (int)(bht % TDIM);
  const int bh = (int)(bht / TDIM);
  const int h  = bh % HHEADS;
  const int b  = bh / HHEADS;

  const unsigned short* qs = qb + bht * (size_t)(NPAD * HD);
  const unsigned short* ks = kb + bht * (size_t)(NPAD * HD);
  const unsigned short* vs = vt + bht * (size_t)(HD * NPAD);
  const float scale = 0.125f;  // 1/sqrt(64)

  for (int rb = wave; rb < 13; rb += 8) {
    // ---- A fragments of q for both K-steps (d 0..31, 32..63) ----
    const unsigned short* qrow = qs + (size_t)(rb * 16 + l16) * HD;
    Frag aq[2];
#pragma unroll
    for (int s2 = 0; s2 < 2; ++s2) {
      aq[s2].q[0] = ldg4(qrow + s2 * 32 + 8 * half);
      aq[s2].q[1] = ldg4(qrow + s2 * 32 + 8 * half + 16);
    }

    // ---- S = q k^T over 14 key tiles ----
    Acc s[14];
#pragma unroll
    for (int ct = 0; ct < 14; ++ct) {
#pragma unroll
      for (int i = 0; i < 8; ++i) s[ct].f[i] = 0.f;
      const unsigned short* krow = ks + (size_t)(ct * 16 + l16) * HD;  // B col = key
#pragma unroll
      for (int s2 = 0; s2 < 2; ++s2) {
        Frag bfr;
        bfr.q[0] = ldg4(krow + s2 * 32 + 8 * half);
        bfr.q[1] = ldg4(krow + s2 * 32 + 8 * half + 16);
        s[ct].v = __builtin_amdgcn_wmma_f32_16x16x32_bf16(false, aq[s2].v, false, bfr.v,
                                                          (short)0, s[ct].v, false, false);
      }
    }

    // ---- softmax over keys: D layout N = l16 -> reduce across 16-lane groups ----
    float rmax[8], rsum[8];
#pragma unroll
    for (int r = 0; r < 8; ++r) rmax[r] = -3.0e38f;
#pragma unroll
    for (int ct = 0; ct < 14; ++ct) {
      bool valid = (ct * 16 + l16) < NTOK;
#pragma unroll
      for (int r = 0; r < 8; ++r) {
        float v = valid ? s[ct].f[r] * scale : -3.0e38f;
        s[ct].f[r] = v;
        rmax[r] = fmaxf(rmax[r], v);
      }
    }
#pragma unroll
    for (int r = 0; r < 8; ++r)
#pragma unroll
      for (int m = 1; m < 16; m <<= 1) rmax[r] = fmaxf(rmax[r], __shfl_xor(rmax[r], m, 32));

#pragma unroll
    for (int r = 0; r < 8; ++r) rsum[r] = 0.f;
#pragma unroll
    for (int ct = 0; ct < 14; ++ct) {
#pragma unroll
      for (int r = 0; r < 8; ++r) {
        float p = __expf(s[ct].f[r] - rmax[r]);
        s[ct].f[r] = p;
        rsum[r] += p;
      }
#pragma unroll
      for (int r = 0; r < 8; ++r) {  // P -> LDS row-major [16][NPAD]
        int row = r + 8 * half;
        lds_p[wave][row * NPAD + ct * 16 + l16] = f2bf(s[ct].f[r]);
      }
    }
#pragma unroll
    for (int r = 0; r < 8; ++r)
#pragma unroll
      for (int m = 1; m < 16; m <<= 1) rsum[r] += __shfl_xor(rsum[r], m, 32);

    // ---- out = P @ v_conv : K = 224 in 7 steps, 4 d-tiles ----
#pragma unroll
    for (int dt = 0; dt < 4; ++dt) {
      Acc o;
#pragma unroll
      for (int i = 0; i < 8; ++i) o.f[i] = 0.f;
      const unsigned short* vrow = vs + (size_t)(dt * 16 + l16) * NPAD;  // B col = d
#pragma unroll
      for (int kt = 0; kt < 7; ++kt) {
        Frag pa, vbf;
        const unsigned short* prow = &lds_p[wave][l16 * NPAD + kt * 32 + 8 * half];
        pa.q[0] = *(const u32x4*)(const void*)prow;
        pa.q[1] = *(const u32x4*)(const void*)(prow + 16);
        vbf.q[0] = ldg4(vrow + kt * 32 + 8 * half);
        vbf.q[1] = ldg4(vrow + kt * 32 + 8 * half + 16);
        o.v = __builtin_amdgcn_wmma_f32_16x16x32_bf16(false, pa.v, false, vbf.v,
                                                      (short)0, o.v, false, false);
      }
      // store normalized rows (bf16) into [B,T,N,C] with c = h*64 + d
#pragma unroll
      for (int r = 0; r < 8; ++r) {
        int row = rb * 16 + r + 8 * half;
        if (row < NTOK) {
          float val = o.f[r] / rsum[r];
          size_t m = (size_t)(b * TDIM + t) * NTOK + row;
          obuf[m * CDIM + h * HD + dt * 16 + l16] = f2bf(val);
        }
      }
    }
  }
}

// =====================================================================
// Kernel 4: output projection + bias (bf16 in, fp32 out).
// Same 2x4 register blocking as qkv_kernel.
// =====================================================================
__global__ __launch_bounds__(256)
void proj_kernel(const unsigned short* __restrict__ obuf,
                 const unsigned short* __restrict__ wb,
                 const float* __restrict__ bias, float* __restrict__ out) {
  const int lane = threadIdx.x & 31;
  const int wave = threadIdx.x >> 5;
  const int half = lane >> 4;
  const int l16  = lane & 15;
  const int mw   = wave & 1;
  const int nw   = wave >> 1;
  const int rm = blockIdx.y * 64 + mw * 32;
  const int rn = blockIdx.x * 256 + nw * 64;

  const unsigned short* ar[2];
  const unsigned short* br[4];
#pragma unroll
  for (int mi = 0; mi < 2; ++mi) ar[mi] = obuf + (size_t)(rm + mi * 16 + l16) * CDIM;
#pragma unroll
  for (int ni = 0; ni < 4; ++ni) br[ni] = wb + (size_t)(rn + ni * 16 + l16) * CDIM;

  Acc acc[2][4];
#pragma unroll
  for (int mi = 0; mi < 2; ++mi)
#pragma unroll
    for (int ni = 0; ni < 4; ++ni)
#pragma unroll
      for (int i = 0; i < 8; ++i) acc[mi][ni].f[i] = 0.f;

  for (int k0 = 0; k0 < CDIM; k0 += 32) {
    Frag af[2], bf[4];
#pragma unroll
    for (int mi = 0; mi < 2; ++mi) {
      af[mi].q[0] = ldg4(ar[mi] + k0 + 8 * half);
      af[mi].q[1] = ldg4(ar[mi] + k0 + 8 * half + 16);
    }
#pragma unroll
    for (int ni = 0; ni < 4; ++ni) {
      bf[ni].q[0] = ldg4(br[ni] + k0 + 8 * half);
      bf[ni].q[1] = ldg4(br[ni] + k0 + 8 * half + 16);
    }
#pragma unroll
    for (int mi = 0; mi < 2; ++mi)
#pragma unroll
      for (int ni = 0; ni < 4; ++ni)
        acc[mi][ni].v = __builtin_amdgcn_wmma_f32_16x16x32_bf16(
            false, af[mi].v, false, bf[ni].v, (short)0, acc[mi][ni].v, false, false);
  }

#pragma unroll
  for (int ni = 0; ni < 4; ++ni) {
    float bv = bias[rn + ni * 16 + l16];
#pragma unroll
    for (int mi = 0; mi < 2; ++mi)
#pragma unroll
      for (int r = 0; r < 8; ++r) {
        size_t m = (size_t)(rm + mi * 16 + r + 8 * half);
        out[m * CDIM + rn + ni * 16 + l16] = acc[mi][ni].f[r] + bv;
      }
  }
}

// =====================================================================
extern "C" void kernel_launch(void* const* d_in, const int* in_sizes, int n_in,
                              void* d_out, int out_size, void* d_ws, size_t ws_size,
                              hipStream_t stream) {
  (void)in_sizes; (void)n_in; (void)out_size; (void)ws_size;
  const float* x      = (const float*)d_in[0];
  const float* w_qkv  = (const float*)d_in[1];
  const float* w_proj = (const float*)d_in[2];
  const float* b_proj = (const float*)d_in[3];
  const float* w_dw   = (const float*)d_in[4];
  const float* b_dw   = (const float*)d_in[5];
  float* out = (float*)d_out;

  // scratch partition (bf16 buffers, 256B aligned)
  const size_t NBHT  = (size_t)BDIM * HHEADS * TDIM;   // 1536
  const size_t SLICE = (size_t)NPAD * HD;              // 14336 elems
  auto aln = [](size_t v) { return (v + 255) & ~(size_t)255; };
  const size_t SZ_X   = aln((size_t)MROWS * CDIM * 2);
  const size_t SZ_WQ  = aln((size_t)OC3 * CDIM * 2);
  const size_t SZ_WP  = aln((size_t)CDIM * CDIM * 2);
  const size_t SZ_BHT = aln(NBHT * SLICE * 2);
  const size_t SZ_O   = aln((size_t)MROWS * CDIM * 2);

  char* p = (char*)d_ws;
  unsigned short* xb   = (unsigned short*)p;  p += SZ_X;
  unsigned short* wqb  = (unsigned short*)p;  p += SZ_WQ;
  unsigned short* wpb  = (unsigned short*)p;  p += SZ_WP;
  unsigned short* qbuf = (unsigned short*)p;  p += SZ_BHT;
  unsigned short* kbuf = (unsigned short*)p;  p += SZ_BHT;
  unsigned short* vbuf = (unsigned short*)p;  p += SZ_BHT;
  unsigned short* vt   = (unsigned short*)p;  p += SZ_BHT;
  unsigned short* obuf = (unsigned short*)p;  p += SZ_O;

  // 0) one-shot fp32 -> bf16 conversions
  {
    int n4 = MROWS * CDIM / 4;
    cvt_kernel<<<dim3((n4 + 255) / 256), dim3(256), 0, stream>>>(x, xb, n4);
    n4 = OC3 * CDIM / 4;
    cvt_kernel<<<dim3((n4 + 255) / 256), dim3(256), 0, stream>>>(w_qkv, wqb, n4);
    n4 = CDIM * CDIM / 4;
    cvt_kernel<<<dim3((n4 + 255) / 256), dim3(256), 0, stream>>>(w_proj, wpb, n4);
  }

  // 1) QKV projection: grid 9 x 392 (256-col x 64-row blocks)
  qkv_kernel<<<dim3(OC3 / 256, MROWS / 64), dim3(256), 0, stream>>>(xb, wqb, qbuf, kbuf, vbuf);

  // 2) depthwise conv + transpose of V
  {
    size_t total = NBHT * (size_t)HD * NPAD;
    int blocks = (int)((total + 255) / 256);
    conv_kernel<<<dim3(blocks), dim3(256), 0, stream>>>(vbuf, w_dw, b_dw, vt);
  }

  // 3) attention: one block per (b,h,t)
  attn_kernel<<<dim3((unsigned)NBHT), dim3(256), 0, stream>>>(qbuf, kbuf, vt, obuf);

  // 4) output projection: grid 3 x 392
  proj_kernel<<<dim3(CDIM / 256, MROWS / 64), dim3(256), 0, stream>>>(obuf, wpb, b_proj, out);
}
